// LIIF_34205119545644
// MI455X (gfx1250) — matrix operand order repaired and checked by reading, not compiled, not run on hardware
//
#include <hip/hip_runtime.h>
#include <hip/hip_bf16.h>

typedef __attribute__((ext_vector_type(16))) _Float16 v16h;
typedef __attribute__((ext_vector_type(8)))  _Float16 v8h;
typedef __attribute__((ext_vector_type(8)))  float    v8f;

#define Bc   4
#define Dc   64
#define Hc   64
#define Wc   64
#define Sc   64
#define Nc   8192
#define Kc   4
#define HW   (Hc*Wc)          // 4096
#define D9   (Dc*9)           // 576
#define HID  256
#define K0PAD 608             // 580 padded to multiple of 32 (19 blocks)
#define RTOT (Bc*Nc*8)        // 262144 MLP rows (4 attn + 4 local per query)
#define MT   64               // rows per MLP block tile

struct Meta { int src; float e0, e1, e2, e3, wgt; };

static __device__ __forceinline__ int iclamp(int v, int lo, int hi) {
    return v < lo ? lo : (v > hi ? hi : v);
}

// ---------------- 1. feat unfold -> f16 row table (B*hw, 576) -------------
__global__ void k_unfold(const float* __restrict__ feat, _Float16* __restrict__ out) {
    int id = blockIdx.x * 256 + threadIdx.x;
    if (id >= Bc * HW * D9) return;
    int c9  = id % D9;
    int pix = id / D9;
    int b = pix >> 12, p = pix & (HW - 1);
    int y = p >> 6, x = p & 63;
    int c  = c9 / 9, kk = c9 % 9;
    int ki = kk / 3, kj = kk % 3;
    int yy = y + ki - 1, xx = x + kj - 1;
    float v = 0.f;
    if (yy >= 0 && yy < Hc && xx >= 0 && xx < Wc)
        v = feat[((size_t)(b * Dc + c) * Hc + yy) * Wc + xx];
    out[(size_t)pix * D9 + c9] = (_Float16)v;
}

// ---------------- 2. weight swizzle into WMMA B-fragment layout -----------
// dst[((ntile*kBlocks + kb)*512) + lane*16 + j]; h=lane/16: j<8: K=8h+j, j>=8: K=16+8h+(j-8)
__global__ void k_swz(const float* __restrict__ w, _Float16* __restrict__ dst,
                      int Ksrc, int kBlocks) {
    int id = blockIdx.x * 256 + threadIdx.x;
    int total = kBlocks * 16 * 512;
    if (id >= total) return;
    int j    = id & 15;
    int lane = (id >> 4) & 31;
    int blk  = id >> 9;
    int kb    = blk % kBlocks;
    int ntile = blk / kBlocks;
    int i = j >> 1, pr = j & 1;
    int kLocal = (j < 8) ? (8 * (lane >> 4) + j) : (16 + 8 * (lane >> 4) + (j - 8));
    int k = kb * 32 + kLocal;
    int n = ntile * 16 + (lane & 15);
    float v = (k < Ksrc) ? w[(size_t)k * HID + n] : 0.f;
    dst[id] = (_Float16)v;
}

// ---------------- 3. top-4 per (b, segment) over attn_t columns -----------
__global__ void k_top4(const float* __restrict__ attn, float* __restrict__ vals,
                       int* __restrict__ idxs) {
    int t = threadIdx.x;           // 256 = B*S
    int b = t >> 6, s = t & 63;
    float v0 = -1e30f, v1 = -1e30f, v2 = -1e30f, v3 = -1e30f;
    int   i0 = 0, i1 = 0, i2 = 0, i3 = 0;
    for (int p = 0; p < HW; ++p) {
        float a = attn[((size_t)b * HW + p) * Sc + s];
        if (a > v0)      { v3=v2;i3=i2; v2=v1;i2=i1; v1=v0;i1=i0; v0=a;i0=p; }
        else if (a > v1) { v3=v2;i3=i2; v2=v1;i2=i1; v1=a;i1=p; }
        else if (a > v2) { v3=v2;i3=i2; v2=a;i2=p; }
        else if (a > v3) { v3=a;i3=p; }
    }
    vals[t*4+0]=v0; vals[t*4+1]=v1; vals[t*4+2]=v2; vals[t*4+3]=v3;
    idxs[t*4+0]=i0; idxs[t*4+1]=i1; idxs[t*4+2]=i2; idxs[t*4+3]=i3;
}

// ---------------- 4a. attention-row metadata ------------------------------
__global__ void k_attn_rows(const float* __restrict__ attn, const float* __restrict__ coord,
                            const float* __restrict__ cell, const float* __restrict__ t4v,
                            const int* __restrict__ t4i, Meta* __restrict__ meta) {
    int id = blockIdx.x * 256 + threadIdx.x;
    if (id >= Bc * Nc) return;
    int b = id >> 13;
    float c0 = coord[(size_t)id*2+0], c1 = coord[(size_t)id*2+1];
    int p0 = iclamp((int)((c0 + 1.f) * 0.5f * 64.f), 0, HW - 1);
    int p1 = iclamp((int)((c1 + 1.f) * 0.5f * 64.f), 0, HW - 1);
    int pc1d = iclamp(p0 * Hc + p1, 0, HW - 1);
    int segm = 0; float best = -1e30f;
    for (int s = 0; s < Sc; ++s) {
        float a = attn[((size_t)b * HW + pc1d) * Sc + s];
        if (a > best) { best = a; segm = s; }
    }
    float rc0 = cell[(size_t)id*2+0] * 64.f, rc1 = cell[(size_t)id*2+1] * 64.f;
    int tb = (b * Sc + segm) * 4;
    float wsum = t4v[tb] + t4v[tb+1] + t4v[tb+2] + t4v[tb+3];
    for (int k = 0; k < Kc; ++k) {
        int rcI = t4i[tb + k];
        int rrc = rcI - pc1d;
        int m = ((rrc % 64) + 64) % 64;
        int q = (rrc - m) / 64;           // floor division
        Meta mt;
        mt.src = rcI;                      // batch-0 flattened gather (faithful)
        mt.e0 = (float)q / 64.f;
        mt.e1 = (float)m / 64.f;
        mt.e2 = rc0; mt.e3 = rc1;
        mt.wgt = 0.5f * t4v[tb + k] / wsum;   // * K/(K+4)
        meta[(size_t)id * 8 + k] = mt;
    }
}

// ---------------- 4b. local-ensemble-row metadata -------------------------
__global__ void k_local_rows(const float* __restrict__ coord, const float* __restrict__ cell,
                             Meta* __restrict__ meta) {
    int id = blockIdx.x * 256 + threadIdx.x;
    if (id >= Bc * Nc) return;
    int b = id >> 13;
    float c0 = coord[(size_t)id*2+0], c1 = coord[(size_t)id*2+1];
    float rc0 = cell[(size_t)id*2+0] * 64.f, rc1 = cell[(size_t)id*2+1] * 64.f;
    const float rx = 1.f / 64.f, eps = 1e-6f;
    float area[4]; Meta mt[4];
    for (int j = 0; j < 4; ++j) {
        float o0 = ((j < 2) ? -1.f : 1.f) * rx + eps;
        float o1 = ((j & 1) ?  1.f : -1.f) * rx + eps;
        float cc0 = fminf(fmaxf(c0 + o0, -1.f + 1e-6f), 1.f - 1e-6f);
        float cc1 = fminf(fmaxf(c1 + o1, -1.f + 1e-6f), 1.f - 1e-6f);
        int iy = iclamp((int)rintf(((cc0 + 1.f) * 64.f - 1.f) * 0.5f), 0, Hc - 1);
        int ix = iclamp((int)rintf(((cc1 + 1.f) * 64.f - 1.f) * 0.5f), 0, Wc - 1);
        float qc0 = -1.f + (2.f * iy + 1.f) / 64.f;
        float qc1 = -1.f + (2.f * ix + 1.f) / 64.f;
        float r0 = (c0 - qc0) * 64.f, r1 = (c1 - qc1) * 64.f;
        area[j] = fabsf(r0 * r1) + 1e-9f;
        mt[j].src = b * HW + iy * Wc + ix;
        mt[j].e0 = r0; mt[j].e1 = r1; mt[j].e2 = rc0; mt[j].e3 = rc1;
    }
    float tot = area[0] + area[1] + area[2] + area[3];
    for (int j = 0; j < 4; ++j) {
        mt[j].wgt = 0.5f * area[3 - j] / tot;   // reversed areas, * 4/(K+4)
        meta[(size_t)id * 8 + 4 + j] = mt[j];
    }
}

// ---------------- 5. fused 4x WMMA MLP + head -----------------------------
// Each wave owns a 2(m) x 4(n) patch of 16x16 tiles: 8 WMMAs per k-step from
// 2 A-fragment loads (4x ds_b128) and 4 B-fragment loads (8x global_b128).
static __device__ __forceinline__ void mlp_layer(
        const _Float16* __restrict__ inBuf, int pitch, int kBlocks,
        const _Float16* __restrict__ wsz, const float* __restrict__ bias,
        _Float16* __restrict__ outBuf, int tid) {
    int wave = tid >> 5, lane = tid & 31;
    int r = lane & 15, sel = lane >> 4;
    int mG = (wave & 1) * 2;        // 2 of 4 m-tiles
    int nG = (wave >> 1) * 4;       // 4 of 16 n-tiles
    v8f acc[2][4];
    #pragma unroll
    for (int mi = 0; mi < 2; ++mi)
        #pragma unroll
        for (int ni = 0; ni < 4; ++ni) acc[mi][ni] = (v8f){};
    const _Float16* aBase0 = inBuf + ((mG + 0) * 16 + r) * pitch + sel * 8;
    const _Float16* aBase1 = inBuf + ((mG + 1) * 16 + r) * pitch + sel * 8;
    for (int kb = 0; kb < kBlocks; ++kb) {
        v16h a[2];
        v8h lo0 = *(const v8h*)(aBase0 + kb * 32);
        v8h hi0 = *(const v8h*)(aBase0 + kb * 32 + 16);
        v8h lo1 = *(const v8h*)(aBase1 + kb * 32);
        v8h hi1 = *(const v8h*)(aBase1 + kb * 32 + 16);
        #pragma unroll
        for (int e = 0; e < 8; ++e) {
            a[0][e] = lo0[e]; a[0][8 + e] = hi0[e];
            a[1][e] = lo1[e]; a[1][8 + e] = hi1[e];
        }
        v16h bfr[4];
        #pragma unroll
        for (int ni = 0; ni < 4; ++ni)
            bfr[ni] = *(const v16h*)(wsz + ((size_t)((nG + ni) * kBlocks + kb)) * 512
                                         + lane * 16);
        #pragma unroll
        for (int ni = 0; ni < 4; ++ni)
            #pragma unroll
            for (int mi = 0; mi < 2; ++mi)
                acc[mi][ni] = __builtin_amdgcn_wmma_f32_16x16x32_f16(
                        false, a[mi], false, bfr[ni], (short)0, acc[mi][ni], false, false);
    }
    #pragma unroll
    for (int ni = 0; ni < 4; ++ni) {
        int nCol = (nG + ni) * 16 + r;
        float bv = bias[nCol];
        #pragma unroll
        for (int mi = 0; mi < 2; ++mi) {
            #pragma unroll
            for (int vg = 0; vg < 8; ++vg) {
                int row = (mG + mi) * 16 + vg + sel * 8;
                float v = acc[mi][ni][vg] + bv;
                v = v > 0.f ? v : 0.f;            // ReLU
                outBuf[row * HID + nCol] = (_Float16)v;
            }
        }
    }
}

__global__ __launch_bounds__(256) void k_mlp(
        const _Float16* __restrict__ featRows, const Meta* __restrict__ meta,
        const _Float16* __restrict__ swz0, const _Float16* __restrict__ swz1,
        const _Float16* __restrict__ swz2, const _Float16* __restrict__ swz3,
        const float* __restrict__ b0, const float* __restrict__ b1,
        const float* __restrict__ b2, const float* __restrict__ b3,
        const float* __restrict__ w4, const float* __restrict__ b4,
        float* __restrict__ pred) {
    __shared__ __align__(16) _Float16 sX[MT * K0PAD];   // 77,824 B
    __shared__ __align__(16) _Float16 sH[MT * HID];     // 32,768 B
    __shared__ int   sSrc[MT];
    __shared__ float sE[MT][4];
    __shared__ float sWgt[MT];
    int tid = threadIdx.x;
    int base = blockIdx.x * MT;
    if (tid < MT) {
        Meta m = meta[base + tid];
        sSrc[tid] = m.src;
        sE[tid][0] = m.e0; sE[tid][1] = m.e1; sE[tid][2] = m.e2; sE[tid][3] = m.e3;
        sWgt[tid] = m.wgt;
    }
    __syncthreads();
    // vectorized gather of the 576 feature halves (72 x v8h per row)
    for (int idx = tid; idx < MT * 72; idx += 256) {
        int rl = idx / 72, c = idx - rl * 72;
        *(v8h*)(sX + rl * K0PAD + c * 8) =
            *(const v8h*)(featRows + (size_t)sSrc[rl] * D9 + c * 8);
    }
    // extras + zero pad (columns 576..607)
    for (int idx = tid; idx < MT * 32; idx += 256) {
        int rl = idx >> 5, k = D9 + (idx & 31);
        sX[rl * K0PAD + k] = (k < D9 + 4) ? (_Float16)sE[rl][k - D9] : (_Float16)0.f;
    }
    __syncthreads();
    mlp_layer(sX, K0PAD, K0PAD / 32, swz0, b0, sH, tid);
    __syncthreads();
    mlp_layer(sH, HID, HID / 32, swz1, b1, sX, tid);
    __syncthreads();
    mlp_layer(sX, HID, HID / 32, swz2, b2, sH, tid);
    __syncthreads();
    mlp_layer(sH, HID, HID / 32, swz3, b3, sX, tid);
    __syncthreads();
    if (tid < MT * 3) {                       // 256 -> 3 head, pre-scaled
        int rl = tid / 3, c = tid - rl * 3;
        float s = b4[c];
        const _Float16* hrow = sX + rl * HID;
        for (int k = 0; k < HID; ++k) s += (float)hrow[k] * w4[(size_t)k * 3 + c];
        pred[(size_t)(base + rl) * 3 + c] = s * sWgt[rl];
    }
}

// ---------------- 6. reduce 8 rows per query ------------------------------
__global__ void k_reduce(const float* __restrict__ pred, float* __restrict__ out) {
    int id = blockIdx.x * 256 + threadIdx.x;
    if (id >= Bc * Nc * 3) return;
    int g = id / 3, c = id - g * 3;
    float s = 0.f;
    for (int r = 0; r < 8; ++r) s += pred[(size_t)(g * 8 + r) * 3 + c];
    out[id] = s;
}

extern "C" void kernel_launch(void* const* d_in, const int* in_sizes, int n_in,
                              void* d_out, int out_size, void* d_ws, size_t ws_size,
                              hipStream_t stream) {
    const float* feat  = (const float*)d_in[0];
    const float* attn  = (const float*)d_in[1];
    const float* coord = (const float*)d_in[2];
    const float* cell  = (const float*)d_in[3];
    const float* w0 = (const float*)d_in[4];  const float* b0 = (const float*)d_in[5];
    const float* w1 = (const float*)d_in[6];  const float* b1 = (const float*)d_in[7];
    const float* w2 = (const float*)d_in[8];  const float* b2 = (const float*)d_in[9];
    const float* w3 = (const float*)d_in[10]; const float* b3 = (const float*)d_in[11];
    const float* w4 = (const float*)d_in[12]; const float* b4 = (const float*)d_in[13];
    float* out = (float*)d_out;

    char* p = (char*)d_ws;
    auto carve = [&](size_t bytes) -> void* {
        void* r = (void*)p; p += (bytes + 255) & ~(size_t)255; return r;
    };
    _Float16* featRows = (_Float16*)carve((size_t)Bc * HW * D9 * 2);   // 18.9 MB
    _Float16* swz0 = (_Float16*)carve((size_t)19 * 16 * 512 * 2);
    _Float16* swz1 = (_Float16*)carve((size_t)8 * 16 * 512 * 2);
    _Float16* swz2 = (_Float16*)carve((size_t)8 * 16 * 512 * 2);
    _Float16* swz3 = (_Float16*)carve((size_t)8 * 16 * 512 * 2);
    float*    t4v  = (float*)carve(256 * 4 * 4);
    int*      t4i  = (int*)carve(256 * 4 * 4);
    Meta*     meta = (Meta*)carve((size_t)RTOT * sizeof(Meta));        // 6.3 MB
    float*    pred = (float*)carve((size_t)RTOT * 3 * 4);              // 3.1 MB

    int nUnf = Bc * HW * D9;
    k_unfold<<<(nUnf + 255) / 256, 256, 0, stream>>>(feat, featRows);
    k_swz<<<(19 * 16 * 512 + 255) / 256, 256, 0, stream>>>(w0, swz0, 580, 19);
    k_swz<<<(8 * 16 * 512 + 255) / 256, 256, 0, stream>>>(w1, swz1, 256, 8);
    k_swz<<<(8 * 16 * 512 + 255) / 256, 256, 0, stream>>>(w2, swz2, 256, 8);
    k_swz<<<(8 * 16 * 512 + 255) / 256, 256, 0, stream>>>(w3, swz3, 256, 8);
    k_top4<<<1, 256, 0, stream>>>(attn, t4v, t4i);
    k_attn_rows<<<(Bc * Nc + 255) / 256, 256, 0, stream>>>(attn, coord, cell, t4v, t4i, meta);
    k_local_rows<<<(Bc * Nc + 255) / 256, 256, 0, stream>>>(coord, cell, meta);
    k_mlp<<<RTOT / MT, 256, 0, stream>>>(featRows, meta, swz0, swz1, swz2, swz3,
                                         b0, b1, b2, b3, w4, b4, pred);
    k_reduce<<<(Bc * Nc * 3 + 255) / 256, 256, 0, stream>>>(pred, out);
}
